// StreamingCodecExportWrapper_39737037422963
// MI455X (gfx1250) — matrix operand order repaired and checked by reading, not compiled, not run on hardware
//
#include <hip/hip_runtime.h>

// ---------------- types / constants ----------------
typedef __attribute__((ext_vector_type(16))) _Float16 v16h;
typedef __attribute__((ext_vector_type(8)))  _Float16 v8h;
typedef __attribute__((ext_vector_type(8)))  float    v8f;

static constexpr int kNQ = 8, kCB = 1024, kH = 1024, kNH = 16, kHD = 64;
static constexpr int kDFF = 4096, kNL = 4, kB = 2, kT = 1024;
static constexpr float kEPS = 1e-6f;

static constexpr int BM = 128, BN = 64, BK = 32;
static constexpr int LDS_STR = 40;  // padded LDS row stride (halves); 80B rows -> conflict-free frag reads

// ---------------- WMMA GEMM:  C[M,N] = A[M,K] * B,  A/B are f16 ----------------
// BT_MODE==1: B supplied transposed, [N][K] row-major (all weights + q*k^T) -> vector staging
// BT_MODE==0: B is [K][N] row-major (only o = P*V, N=64) -> scatter staging
// OUT_HALF : 0 -> f32 C, 1 -> f16 C
// Batched over grid.z: z -> (zb=z/hDiv, zh=z%hDiv) with independent element strides.
// Requirements: M%128==0, N%64==0, K%32==0 (all call sites comply).
template <int BT_MODE, int OUT_HALF>
__launch_bounds__(256)
__global__ void gemm_f16wmma(const _Float16* __restrict__ A, const _Float16* __restrict__ B,
                             void* __restrict__ Cv, int M, int N, int K,
                             int lda, int ldb, int ldc, int hDiv,
                             long long aBs, long long aHs,
                             long long bBs, long long bHs,
                             long long cBs, long long cHs) {
  __shared__ _Float16 As[2][BM * LDS_STR];
  __shared__ _Float16 Bs[2][BN * LDS_STR];

  const int tid = threadIdx.x;
  const int z   = blockIdx.z;
  const long long zb = z / hDiv, zh = z % hDiv;
  A += zb * aBs + zh * aHs;
  B += zb * bBs + zh * bHs;
  const long long coff = zb * cBs + zh * cHs;

  const int lane  = tid & 31;
  const int wave  = tid >> 5;
  const int waveM = wave & 3;   // 4 waves over M (32 rows each)
  const int waveN = wave >> 2;  // 2 waves over N (32 cols each)
  const int half  = lane >> 4;  // K-half selector (ISA wave32 layout)
  const int l16   = lane & 15;

  const int rowBase = blockIdx.y * BM;
  const int colBase = blockIdx.x * BN;

  // register relay for double buffering
  v8h aR[2], bR;
  auto loadA = [&](int k0) {
#pragma unroll
    for (int i = 0; i < 2; ++i) {
      int idx8 = tid + i * 256;          // 512 x v8h slots = 128x32
      int r = idx8 >> 2, c8 = idx8 & 3;
      aR[i] = *(const v8h*)&A[(size_t)(rowBase + r) * lda + k0 + c8 * 8];
    }
  };
  auto storeA = [&](int buf) {
#pragma unroll
    for (int i = 0; i < 2; ++i) {
      int idx8 = tid + i * 256;
      int r = idx8 >> 2, c8 = idx8 & 3;
      *(v8h*)&As[buf][r * LDS_STR + c8 * 8] = aR[i];
    }
  };
  auto loadB = [&](int k0) {
    if (BT_MODE) {
      int r = tid >> 2, c8 = tid & 3;    // 256 x v8h slots = 64x32
      bR = *(const v8h*)&B[(size_t)(colBase + r) * ldb + k0 + c8 * 8];
    } else {
      int k = tid >> 3, n8 = tid & 7;    // 32 x 64 tile, 8 halves along n
      bR = *(const v8h*)&B[(size_t)(k0 + k) * ldb + colBase + n8 * 8];
    }
  };
  auto storeB = [&](int buf) {
    if (BT_MODE) {
      int r = tid >> 2, c8 = tid & 3;
      *(v8h*)&Bs[buf][r * LDS_STR + c8 * 8] = bR;
    } else {
      int k = tid >> 3, n8 = tid & 7;
#pragma unroll
      for (int e = 0; e < 8; ++e) Bs[buf][(n8 * 8 + e) * LDS_STR + k] = bR[e];
    }
  };

  v8f acc[2][2];
#pragma unroll
  for (int i = 0; i < 2; ++i)
#pragma unroll
    for (int j = 0; j < 2; ++j) acc[i][j] = {};

  const int nt = K / BK;
  loadA(0); loadB(0);
  storeA(0); storeB(0);
  __syncthreads();

  for (int t = 0; t < nt; ++t) {
    const int cur = t & 1;
    const bool more = (t + 1 < nt);
    if (more) { loadA((t + 1) * BK); loadB((t + 1) * BK); }  // in flight during compute

    // fragments per ISA 7.12.2 wave32 layouts
    v16h af[2], bf[2];
#pragma unroll
    for (int i = 0; i < 2; ++i) {
      int row = waveM * 32 + i * 16 + l16;
      v8h lo = *(const v8h*)&As[cur][row * LDS_STR + half * 8];       // K = half*8+0..7
      v8h hi = *(const v8h*)&As[cur][row * LDS_STR + 16 + half * 8];  // K = 16+half*8+0..7
      af[i] = __builtin_shufflevector(lo, hi, 0,1,2,3,4,5,6,7,8,9,10,11,12,13,14,15);
    }
#pragma unroll
    for (int j = 0; j < 2; ++j) {
      int n = waveN * 32 + j * 16 + l16;
      v8h lo = *(const v8h*)&Bs[cur][n * LDS_STR + half * 8];
      v8h hi = *(const v8h*)&Bs[cur][n * LDS_STR + 16 + half * 8];
      bf[j] = __builtin_shufflevector(lo, hi, 0,1,2,3,4,5,6,7,8,9,10,11,12,13,14,15);
    }
    if (more) { storeA(1 - cur); storeB(1 - cur); }  // writes go to the other buffer
#pragma unroll
    for (int i = 0; i < 2; ++i)
#pragma unroll
      for (int j = 0; j < 2; ++j)
        acc[i][j] = __builtin_amdgcn_wmma_f32_16x16x32_f16(
            false, af[i], false, bf[j], (short)0, acc[i][j], false, false);
    __syncthreads();
  }

  // store C: VGPR r holds M = r + half*8, N = lane&15
#pragma unroll
  for (int i = 0; i < 2; ++i)
#pragma unroll
    for (int j = 0; j < 2; ++j) {
      int n  = colBase + waveN * 32 + j * 16 + l16;
      int rb = rowBase + waveM * 32 + i * 16 + half * 8;
      if (OUT_HALF) {
        _Float16* C = (_Float16*)Cv + coff;
#pragma unroll
        for (int r = 0; r < 8; ++r) C[(size_t)(rb + r) * ldc + n] = (_Float16)acc[i][j][r];
      } else {
        float* C = (float*)Cv + coff;
#pragma unroll
        for (int r = 0; r < 8; ++r) C[(size_t)(rb + r) * ldc + n] = acc[i][j][r];
      }
    }
}

// ---------------- weight convert+transpose: f32 [K][N] -> f16 [N][K], grid.z = layer ----------------
__global__ void convT_kernel(const float* __restrict__ in, _Float16* __restrict__ out, int K, int N) {
  long long base = (long long)blockIdx.z * K * N;
  int idx = blockIdx.x * 256 + threadIdx.x;
  if (idx >= K * N) return;
  int k = idx % K, n = idx / K;
  out[base + idx] = (_Float16)in[base + (size_t)k * N + n];
}

// ---------------- RVQ embedding sum -> channel-first [B][H][T] ----------------
__global__ void emb_kernel(const int* __restrict__ codes, const float* __restrict__ cb,
                           float* __restrict__ outCF, int total) {
  int idx = blockIdx.x * 256 + threadIdx.x;
  if (idx >= total) return;
  int t = idx % kT;
  int h = (idx / kT) % kH;
  int b = idx / (kT * kH);
  float s = 0.f;
#pragma unroll
  for (int q = 0; q < kNQ; ++q) {
    int c = codes[(b * kNQ + q) * kT + t];
    s += cb[((size_t)q * kCB + c) * kH + h];
  }
  outCF[idx] = s;
}

// ---------------- generic conv1d over [B][C][L] (handles lhs-dilated transposed-conv) ----------------
__global__ void conv1d_kernel(const float* __restrict__ x, const float* __restrict__ w,
                              const float* __restrict__ bias, const float* __restrict__ resid,
                              float* __restrict__ y, int Bn, int Cin, int Cout,
                              int Lin, int Lout, int Kk, int pad, int dil, int flip, int leaky) {
  int idx = blockIdx.x * 256 + threadIdx.x;
  int total = Bn * Cout * Lout;
  if (idx >= total) return;
  int t  = idx % Lout;
  int co = (idx / Lout) % Cout;
  int b  = idx / (Lout * Cout);
  int kk0 = (pad - t) % dil;
  if (kk0 < 0) kk0 += dil;
  float acc = bias[co];
  const float* xb = x + (size_t)b * Cin * Lin;
  const float* wc = w + (size_t)co * Cin * Kk;
  for (int kk = kk0; kk < Kk; kk += dil) {
    int p = t + kk - pad;
    if (p < 0) continue;
    int ip = p / dil;
    if (ip >= Lin) continue;
    int widx = flip ? (Kk - 1 - kk) : kk;
    for (int ci = 0; ci < Cin; ++ci) {
      float xv = xb[(size_t)ci * Lin + ip];
      if (leaky) xv = (xv < 0.f) ? 0.1f * xv : xv;
      acc += wc[(size_t)ci * Kk + widx] * xv;
    }
  }
  if (resid) acc += resid[idx];
  y[idx] = acc;
}

// ---------------- layout transposes ----------------
__global__ void cf2tf_kernel(const float* __restrict__ in, float* __restrict__ out,
                             int Bn, int C, int L) {  // [B][C][L] -> [B][L][C]
  int idx = blockIdx.x * 256 + threadIdx.x;
  if (idx >= Bn * C * L) return;
  int c = idx % C;
  int l = (idx / C) % L;
  int b = idx / (C * L);
  out[idx] = in[((size_t)b * C + c) * L + l];
}
__global__ void tf2cf_kernel(const float* __restrict__ in, float* __restrict__ out,
                             int Bn, int C, int L) {  // [B][L][C] -> [B][C][L]
  int idx = blockIdx.x * 256 + threadIdx.x;
  if (idx >= Bn * C * L) return;
  int l = idx % L;
  int c = (idx / L) % C;
  int b = idx / (C * L);
  out[idx] = in[((size_t)b * L + l) * C + c];
}

// ---------------- RMSNorm: one block per row ----------------
__global__ void rmsnorm_f16_kernel(const float* __restrict__ x, const float* __restrict__ g,
                                   _Float16* __restrict__ y, int H) {
  int row = blockIdx.x, tid = threadIdx.x;
  const float* xr = x + (size_t)row * H;
  float ss = 0.f;
  for (int i = tid; i < H; i += 256) { float v = xr[i]; ss += v * v; }
  __shared__ float red[256];
  red[tid] = ss; __syncthreads();
  for (int s = 128; s > 0; s >>= 1) { if (tid < s) red[tid] += red[tid + s]; __syncthreads(); }
  float inv = rsqrtf(red[0] / H + kEPS);
  for (int i = tid; i < H; i += 256) y[(size_t)row * H + i] = (_Float16)(xr[i] * inv * g[i]);
}
__global__ void rmsnorm_f32_kernel(const float* __restrict__ x, const float* __restrict__ g,
                                   float* __restrict__ y, int H) {
  int row = blockIdx.x, tid = threadIdx.x;
  const float* xr = x + (size_t)row * H;
  float ss = 0.f;
  for (int i = tid; i < H; i += 256) { float v = xr[i]; ss += v * v; }
  __shared__ float red[256];
  red[tid] = ss; __syncthreads();
  for (int s = 128; s > 0; s >>= 1) { if (tid < s) red[tid] += red[tid + s]; __syncthreads(); }
  float inv = rsqrtf(red[0] / H + kEPS);
  for (int i = tid; i < H; i += 256) y[(size_t)row * H + i] = xr[i] * inv * g[i];
}

// ---------------- causal softmax: f32 scores in, f16 probabilities out ----------------
__global__ void softmax_causal_kernel(float* __restrict__ s, _Float16* __restrict__ p,
                                      int T, float scale) {
  int q = blockIdx.x, bh = blockIdx.y, tid = threadIdx.x;
  float*     row  = s + ((size_t)bh * T + q) * T;
  _Float16*  prow = p + ((size_t)bh * T + q) * T;
  int valid = q + 1;
  __shared__ float red[256];
  float m = -3.4e38f;
  for (int k = tid; k < valid; k += 256) m = fmaxf(m, row[k] * scale);
  red[tid] = m; __syncthreads();
  for (int st = 128; st > 0; st >>= 1) { if (tid < st) red[tid] = fmaxf(red[tid], red[tid + st]); __syncthreads(); }
  m = red[0]; __syncthreads();
  float ls = 0.f;
  for (int k = tid; k < valid; k += 256) { float e = __expf(row[k] * scale - m); row[k] = e; ls += e; }
  red[tid] = ls; __syncthreads();
  for (int st = 128; st > 0; st >>= 1) { if (tid < st) red[tid] += red[tid + st]; __syncthreads(); }
  float inv = 1.f / red[0];
  for (int k = tid; k < T; k += 256)
    prow[k] = (_Float16)((k < valid) ? row[k] * inv : 0.f);
}

// ---------------- elementwise ----------------
__global__ void add_inplace_kernel(float* __restrict__ x, const float* __restrict__ d, int n) {
  int i = blockIdx.x * 256 + threadIdx.x;
  if (i < n) x[i] += d[i];
}
__global__ void silu_mul_f16_kernel(const float* __restrict__ g, const float* __restrict__ u,
                                    _Float16* __restrict__ out, int n) {
  int i = blockIdx.x * 256 + threadIdx.x;
  if (i < n) { float v = g[i]; out[i] = (_Float16)(v * (1.f / (1.f + __expf(-v))) * u[i]); }
}
__global__ void clip_kernel(const float* __restrict__ y, float* __restrict__ out, int n) {
  int i = blockIdx.x * 256 + threadIdx.x;
  if (i < n) out[i] = fminf(fmaxf(y[i], -1.f), 1.f);
}

// ---------------- host orchestration ----------------
static inline int cdiv(long long a, long long b) { return (int)((a + b - 1) / b); }

extern "C" void kernel_launch(void* const* d_in, const int* in_sizes, int n_in,
                              void* d_out, int out_size, void* d_ws, size_t ws_size,
                              hipStream_t stream) {
  (void)in_sizes; (void)n_in; (void)out_size; (void)ws_size;
  const int*   codes = (const int*)d_in[0];
  const float* cb    = (const float*)d_in[1];
  const float* pre_w = (const float*)d_in[2];
  const float* pre_b = (const float*)d_in[3];
  const float* wq    = (const float*)d_in[4];
  const float* wk    = (const float*)d_in[5];
  const float* wv    = (const float*)d_in[6];
  const float* wo    = (const float*)d_in[7];
  const float* wg    = (const float*)d_in[8];
  const float* wu    = (const float*)d_in[9];
  const float* wd    = (const float*)d_in[10];
  const float* ln1   = (const float*)d_in[11];
  const float* ln2   = (const float*)d_in[12];
  const float* lnf   = (const float*)d_in[13];
  const float* upw[4] = {(const float*)d_in[14], (const float*)d_in[16],
                         (const float*)d_in[18], (const float*)d_in[20]};
  const float* upb[4] = {(const float*)d_in[15], (const float*)d_in[17],
                         (const float*)d_in[19], (const float*)d_in[21]};
  const float* dec0w = (const float*)d_in[22];
  const float* dec0b = (const float*)d_in[23];
  const float* dec1w = (const float*)d_in[24];
  const float* dec1b = (const float*)d_in[25];
  const float* outw  = (const float*)d_in[26];
  const float* outb  = (const float*)d_in[27];
  float* out = (float*)d_out;

  // ---- workspace arena (~564 MiB) ----
  char* ws = (char*)d_ws;
  const size_t MiB = 1ull << 20;
  float*    bigA  = (float*)(ws + 0 * MiB);     // 128 MiB; doubles as f32 scores
  float*    bigB  = (float*)(ws + 128 * MiB);   // 128 MiB
  float*    xbuf  = (float*)(ws + 256 * MiB);   // 8 MiB  [B][T][H] residual stream
  float*    t2    = (float*)(ws + 264 * MiB);   // 8 MiB  f32 GEMM outputs
  float*    gate  = (float*)(ws + 272 * MiB);   // 32 MiB
  float*    up    = (float*)(ws + 304 * MiB);   // 32 MiB
  _Float16* t1h   = (_Float16*)(ws + 336 * MiB);  // 4 MiB
  _Float16* qh    = (_Float16*)(ws + 340 * MiB);  // 4 MiB
  _Float16* kh    = (_Float16*)(ws + 344 * MiB);  // 4 MiB
  _Float16* vh    = (_Float16*)(ws + 348 * MiB);  // 4 MiB
  _Float16* oh    = (_Float16*)(ws + 352 * MiB);  // 4 MiB
  _Float16* gateh = (_Float16*)(ws + 356 * MiB);  // 16 MiB
  _Float16* pbuf  = (_Float16*)(ws + 372 * MiB);  // 64 MiB f16 probs
  _Float16* wqT   = (_Float16*)(ws + 436 * MiB);  // 8 MiB (NL x [N][K])
  _Float16* wkT   = (_Float16*)(ws + 444 * MiB);
  _Float16* wvT   = (_Float16*)(ws + 452 * MiB);
  _Float16* woT   = (_Float16*)(ws + 460 * MiB);
  _Float16* wgT   = (_Float16*)(ws + 468 * MiB);  // 32 MiB
  _Float16* wuT   = (_Float16*)(ws + 500 * MiB);  // 32 MiB
  _Float16* wdT   = (_Float16*)(ws + 532 * MiB);  // 32 MiB
  float* scores = bigA;

  const int BT = kB * kT;          // 2048
  const int nBTH = BT * kH;        // 2 Mi
  const float scale = 0.125f;      // 1/sqrt(64)
  const long long HH = (long long)kH * kH, HD4 = (long long)kH * kDFF;

  // 0) weights -> f16 transposed [N][K], per layer via grid.z
  convT_kernel<<<dim3(cdiv(HH, 256), 1, kNL), 256, 0, stream>>>(wq, wqT, kH, kH);
  convT_kernel<<<dim3(cdiv(HH, 256), 1, kNL), 256, 0, stream>>>(wk, wkT, kH, kH);
  convT_kernel<<<dim3(cdiv(HH, 256), 1, kNL), 256, 0, stream>>>(wv, wvT, kH, kH);
  convT_kernel<<<dim3(cdiv(HH, 256), 1, kNL), 256, 0, stream>>>(wo, woT, kH, kH);
  convT_kernel<<<dim3(cdiv(HD4, 256), 1, kNL), 256, 0, stream>>>(wg, wgT, kH, kDFF);
  convT_kernel<<<dim3(cdiv(HD4, 256), 1, kNL), 256, 0, stream>>>(wu, wuT, kH, kDFF);
  convT_kernel<<<dim3(cdiv(HD4, 256), 1, kNL), 256, 0, stream>>>(wd, wdT, kDFF, kH);

  // 1) RVQ embedding -> [B][H][T];  2) pre_conv;  3) -> [B][T][H]
  emb_kernel<<<cdiv(nBTH, 256), 256, 0, stream>>>(codes, cb, bigB, nBTH);
  conv1d_kernel<<<cdiv(nBTH, 256), 256, 0, stream>>>(bigB, pre_w, pre_b, nullptr, bigA,
                                                     kB, kH, kH, kT, kT, 3, 1, 1, 0, 0);
  cf2tf_kernel<<<cdiv(nBTH, 256), 256, 0, stream>>>(bigA, xbuf, kB, kH, kT);

  // 4) transformer layers
  for (int l = 0; l < kNL; ++l) {
    rmsnorm_f16_kernel<<<BT, 256, 0, stream>>>(xbuf, ln1 + (size_t)l * kH, t1h, kH);

    gemm_f16wmma<1, 1><<<dim3(kH / BN, BT / BM, 1), 256, 0, stream>>>(
        t1h, wqT + l * HH, qh, BT, kH, kH, kH, kH, kH, 1, 0, 0, 0, 0, 0, 0);
    gemm_f16wmma<1, 1><<<dim3(kH / BN, BT / BM, 1), 256, 0, stream>>>(
        t1h, wkT + l * HH, kh, BT, kH, kH, kH, kH, kH, 1, 0, 0, 0, 0, 0, 0);
    gemm_f16wmma<1, 1><<<dim3(kH / BN, BT / BM, 1), 256, 0, stream>>>(
        t1h, wvT + l * HH, vh, BT, kH, kH, kH, kH, kH, 1, 0, 0, 0, 0, 0, 0);

    // scores[bh] = q_bh (Tx64) * k_bh^T   (k rows == B^T rows)
    gemm_f16wmma<1, 0><<<dim3(kT / BN, kT / BM, kB * kNH), 256, 0, stream>>>(
        qh, kh, scores, kT, kT, kHD, kH, kH, kT, kNH,
        (long long)kT * kH, (long long)kHD,
        (long long)kT * kH, (long long)kHD,
        (long long)kNH * kT * kT, (long long)kT * kT);

    softmax_causal_kernel<<<dim3(kT, kB * kNH), 256, 0, stream>>>(scores, pbuf, kT, scale);

    // o[bh] = P_bh (TxT) * v_bh (Tx64)   (normal-B path)
    gemm_f16wmma<0, 1><<<dim3(kHD / BN, kT / BM, kB * kNH), 256, 0, stream>>>(
        pbuf, vh, oh, kT, kHD, kT, kT, kH, kH, kNH,
        (long long)kNH * kT * kT, (long long)kT * kT,
        (long long)kT * kH, (long long)kHD,
        (long long)kT * kH, (long long)kHD);

    gemm_f16wmma<1, 0><<<dim3(kH / BN, BT / BM, 1), 256, 0, stream>>>(
        oh, woT + l * HH, t2, BT, kH, kH, kH, kH, kH, 1, 0, 0, 0, 0, 0, 0);
    add_inplace_kernel<<<cdiv(nBTH, 256), 256, 0, stream>>>(xbuf, t2, nBTH);

    rmsnorm_f16_kernel<<<BT, 256, 0, stream>>>(xbuf, ln2 + (size_t)l * kH, t1h, kH);
    gemm_f16wmma<1, 0><<<dim3(kDFF / BN, BT / BM, 1), 256, 0, stream>>>(
        t1h, wgT + l * HD4, gate, BT, kDFF, kH, kH, kH, kDFF, 1, 0, 0, 0, 0, 0, 0);
    gemm_f16wmma<1, 0><<<dim3(kDFF / BN, BT / BM, 1), 256, 0, stream>>>(
        t1h, wuT + l * HD4, up, BT, kDFF, kH, kH, kH, kDFF, 1, 0, 0, 0, 0, 0, 0);
    silu_mul_f16_kernel<<<cdiv((long long)BT * kDFF, 256), 256, 0, stream>>>(gate, up, gateh, BT * kDFF);
    gemm_f16wmma<1, 0><<<dim3(kH / BN, BT / BM, 1), 256, 0, stream>>>(
        gateh, wdT + l * HD4, t2, BT, kH, kDFF, kDFF, kDFF, kH, 1, 0, 0, 0, 0, 0, 0);
    add_inplace_kernel<<<cdiv(nBTH, 256), 256, 0, stream>>>(xbuf, t2, nBTH);
  }

  // 5) final norm, back to channel-first
  rmsnorm_f32_kernel<<<BT, 256, 0, stream>>>(xbuf, lnf, t2, kH);
  tf2cf_kernel<<<cdiv(nBTH, 256), 256, 0, stream>>>(t2, bigA, kB, kH, kT);

  // 6) transposed-conv upsampling (flip=1, leaky input, lhs-dilation = ratio)
  const int ratios[4] = {8, 4, 4, 2};
  const int chans[5]  = {1024, 512, 256, 128, 64};
  float* cur = bigA;
  float* nxt = bigB;
  int Lin = kT;
  for (int i = 0; i < 4; ++i) {
    int r = ratios[i], K = 2 * r;
    int Cin = chans[i], Cout = chans[i + 1];
    int pad = K - 1 - r / 2;
    int Lout = Lin * r;
    long long total = (long long)kB * Cout * Lout;
    conv1d_kernel<<<cdiv(total, 256), 256, 0, stream>>>(cur, upw[i], upb[i], nullptr, nxt,
                                                        kB, Cin, Cout, Lin, Lout, K, pad, r, 1, 1);
    float* tmp = cur; cur = nxt; nxt = tmp;
    Lin = Lout;
  }

  // 7) residual decoder blocks (C=64, K=7) + output conv + clip
  const int C = 64, L = Lin;  // 262144
  long long totC = (long long)kB * C * L;
  conv1d_kernel<<<cdiv(totC, 256), 256, 0, stream>>>(cur, dec0w, dec0b, cur, nxt,
                                                     kB, C, C, L, L, 7, 3, 1, 0, 1);
  { float* tmp = cur; cur = nxt; nxt = tmp; }
  conv1d_kernel<<<cdiv(totC, 256), 256, 0, stream>>>(cur, dec1w, dec1b, cur, nxt,
                                                     kB, C, C, L, L, 7, 3, 1, 0, 1);
  { float* tmp = cur; cur = nxt; nxt = tmp; }
  long long totO = (long long)kB * 1 * L;
  conv1d_kernel<<<cdiv(totO, 256), 256, 0, stream>>>(cur, outw, outb, nullptr, nxt,
                                                     kB, C, 1, L, L, 7, 3, 1, 0, 0);
  clip_kernel<<<cdiv(totO, 256), 256, 0, stream>>>(nxt, out, (int)totO);
}
